// density_wfa_finetune_11072425689457
// MI455X (gfx1250) — matrix operand order repaired and checked by reading, not compiled, not run on hardware
//
#include <hip/hip_runtime.h>
#include <hip/hip_bf16.h>
#include <math.h>

#define NB    8192
#define TT    256
#define XDIM  8
#define EDIM  128
#define RDIM  128
#define NHDIM 128
#define KMIX  8
#define F_LOG2PI 1.8378770664093453f

#define MTILE 64          // rows per block (4 wave32 row-groups)
#define SPAD  68          // padded row stride of transposed S tile (16B aligned, bank-spread)

typedef __attribute__((ext_vector_type(16))) __bf16 v16bf;
typedef __attribute__((ext_vector_type(8)))  float  v8f;

struct __align__(16) U4 { unsigned int x, y, z, w; };
struct B32 { U4 lo, hi; };

// ---- CDNA5 async global->LDS copy (ASYNCcnt-tracked), per ISA 15.18.3 ----
__device__ __forceinline__ void async_copy_b128(unsigned lds_off, const void* gptr) {
  asm volatile("global_load_async_to_lds_b128 %0, %1, off"
               :: "v"(lds_off), "v"(gptr) : "memory");
}
__device__ __forceinline__ void wait_async0() {
  asm volatile("s_wait_asynccnt 0x0" ::: "memory");
}
__device__ __forceinline__ unsigned lds_addr(const void* p) {
  return (unsigned)(unsigned long long)p;   // low 32 bits of LDS aperture = LDS byte address
}

// ---------------- init: S0 = tile(init_w), out = 0 ----------------
__global__ void init_kernel(const float* __restrict__ init_w,
                            float* __restrict__ S0,
                            float* __restrict__ out /* rs[N] ++ ns[N] */)
{
  int idx = blockIdx.x * blockDim.x + threadIdx.x;
  if (idx < NB * RDIM) S0[idx] = init_w[idx & (RDIM - 1)];
  if (idx < 2 * NB)    out[idx] = 0.f;
}

// ------------- A[i,d,j] (f32) -> At[i][j][d] (bf16, d contiguous) -------------
__global__ void transposeA_kernel(const float* __restrict__ A, __bf16* __restrict__ At)
{
  int i = blockIdx.x >> 7;
  int d = blockIdx.x & 127;
  int j = threadIdx.x;
  float v = A[((size_t)(i * EDIM) + d) * RDIM + j];
  At[((size_t)(i * RDIM + j)) * EDIM + d] = (__bf16)v;
}

// ---------------- per-step scalar head: encoder, NADE/GMM, rs/ns ----------------
struct StepParams {
  const float *X, *e1w, *e1b, *e2w, *e2b, *nw, *nbv, *mw, *mb, *sw, *sb, *aw, *ab;
  const float* S;
  __bf16* Ebf;
  float* rs; float* ns;
  int t;
};

__global__ __launch_bounds__(128) void step_kernel(StepParams p)
{
  const int n = blockIdx.x;
  const int j = threadIdx.x;

  __shared__ float xsh[XDIM];
  __shared__ float h1[EDIM];
  __shared__ float ssh[RDIM];
  __shared__ float gsh[NHDIM];
  __shared__ float mush[KMIX * XDIM];
  __shared__ float lsig[KMIX * XDIM];
  __shared__ float al[KMIX];
  __shared__ float lp[KMIX];
  __shared__ float red[128];

  if (j < XDIM) xsh[j] = p.X[((size_t)n * XDIM + j) * TT + p.t];
  float sv = p.S[(size_t)n * RDIM + j];
  ssh[j] = sv;
  red[j] = sv * sv;
  __syncthreads();

  // h1 = tanh(x @ enc1_w + b)
  float a1 = p.e1b[j];
  #pragma unroll
  for (int k = 0; k < XDIM; ++k) a1 += xsh[k] * p.e1w[k * EDIM + j];
  h1[j] = tanhf(a1);
  __syncthreads();

  // e = tanh(h1 @ enc2_w + b)  -> bf16 for the WMMA update
  float a2 = p.e2b[j];
  for (int k = 0; k < EDIM; ++k) a2 += h1[k] * p.e2w[k * EDIM + j];
  float e = tanhf(a2);
  p.Ebf[(size_t)n * EDIM + j] = (__bf16)e;

  // g = tanh(s @ nade_w + b)
  float a3 = p.nbv[j];
  for (int k = 0; k < RDIM; ++k) a3 += ssh[k] * p.nw[k * NHDIM + j];
  gsh[j] = tanhf(a3);
  __syncthreads();

  if (j < KMIX * XDIM) {               // threads 0..63: mu and log-sigma heads
    float am = p.mb[j], as = p.sb[j];
    for (int k = 0; k < NHDIM; ++k) {
      float g = gsh[k];
      am += g * p.mw[k * (KMIX * XDIM) + j];
      as += g * p.sw[k * (KMIX * XDIM) + j];
    }
    mush[j] = am; lsig[j] = as;        // sig = exp(lsig), log(sig) = lsig
  } else if (j < 64 + KMIX) {          // threads 64..71: alpha logits
    int kk = j - 64;
    float aa = p.ab[kk];
    for (int k = 0; k < NHDIM; ++k) aa += gsh[k] * p.aw[k * KMIX + kk];
    al[kk] = aa;
  }
  __syncthreads();

  if (j < KMIX) {                      // component log-likelihoods
    float s = 0.f;
    #pragma unroll
    for (int d = 0; d < XDIM; ++d) {
      float ls = lsig[j * XDIM + d];
      float z  = (xsh[d] - mush[j * XDIM + d]) * expf(-ls);
      s += -0.5f * (z * z + F_LOG2PI) - ls;
    }
    lp[j] = s;
  }
  __syncthreads();
  for (int off = 64; off > 0; off >>= 1) {     // ||s||^2 reduction
    if (j < off) red[j] += red[j + off];
    __syncthreads();
  }
  if (j == 0) {
    float mx = al[0];
    for (int k = 1; k < KMIX; ++k) mx = fmaxf(mx, al[k]);
    float se = 0.f;
    for (int k = 0; k < KMIX; ++k) se += expf(al[k] - mx);
    float lz = mx + logf(se);                  // log softmax denominator
    float vv[KMIX]; float mx2 = -1e30f;
    for (int k = 0; k < KMIX; ++k) { vv[k] = al[k] - lz + lp[k]; mx2 = fmaxf(mx2, vv[k]); }
    float s2 = 0.f;
    for (int k = 0; k < KMIX; ++k) s2 += expf(vv[k] - mx2);
    p.rs[n] += mx2 + logf(s2);
    p.ns[n] += red[0];
  }
}

// ---------------- bilinear state update via WMMA bf16 ----------------
// Block: 128 threads = 4 wave32s. Block owns 64 rows x one 16-wide j-tile.
// Wave w owns row-group w (16 rows). The per-i B slice (16j x 128d, 4KB,
// contiguous in At) is async-copied to LDS once and reused by all 4 waves.
__global__ __launch_bounds__(128) void update_kernel(
    const __bf16* __restrict__ Ebf, const float* __restrict__ Sf,
    const __bf16* __restrict__ At,  float* __restrict__ Snext)
{
  __shared__ __align__(16) __bf16 Esh[MTILE * EDIM];     // 16 KB
  __shared__ __align__(16) float  SshT[RDIM * SPAD];     // S transposed [i][row], 34 KB
  __shared__ __align__(16) __bf16 Bsh[2][16 * EDIM];     // double-buffered B slice, 8 KB

  const int tid  = threadIdx.x;
  const int wave = tid >> 5, lane = tid & 31;
  const int rowBase = blockIdx.x * MTILE;
  const int jb      = blockIdx.y * 16;

  // Stage E tile (bf16) and transposed S tile (f32, padded stride).
  for (int idx = tid; idx < MTILE * EDIM; idx += 128) {
    Esh[idx] = Ebf[(size_t)rowBase * EDIM + idx];
    int r = idx >> 7, i = idx & 127;                     // coalesced global read
    SshT[i * SPAD + r] = Sf[(size_t)(rowBase + r) * RDIM + i];
  }
  // Preload B slice for i = 0 (async global -> LDS).
  {
    const __bf16* g0 = At + ((size_t)jb) * EDIM + (size_t)tid * 16;
    async_copy_b128(lds_addr(&Bsh[0][tid * 16]),     (const void*)g0);
    async_copy_b128(lds_addr(&Bsh[0][tid * 16 + 8]), (const void*)(g0 + 8));
  }
  wait_async0();
  __syncthreads();

  const int M    = lane & 15;
  const int hi   = lane >> 4;

  // Hoisted WMMA A-operands: E rows of this wave's row-group. ISA 16-bit A layout:
  // lanes 0-15 hold K = {0..7, 16..23}; lanes 16-31 hold K = {8..15, 24..31}.
  v16bf aop[4];
  #pragma unroll
  for (int kc = 0; kc < 4; ++kc) {
    const __bf16* p = Esh + (wave * 16 + M) * EDIM + kc * 32 + hi * 8;
    B32 tmp; tmp.lo = *(const U4*)p; tmp.hi = *(const U4*)(p + 16);
    aop[kc] = __builtin_bit_cast(v16bf, tmp);
  }

  float cacc[8];
  #pragma unroll
  for (int r = 0; r < 8; ++r) cacc[r] = 0.f;

  const size_t sliceStride = (size_t)RDIM * EDIM;        // 16384 elements per i
  const __bf16* Bcol = At + (size_t)jb * EDIM;

  for (int i = 0; i < RDIM; ++i) {
    const int cur = i & 1, nxt = cur ^ 1;
    // Kick off async stage of the next B slice; prefetch i+2 into L2.
    if (i + 1 < RDIM) {
      const __bf16* gn = Bcol + (size_t)(i + 1) * sliceStride + (size_t)tid * 16;
      async_copy_b128(lds_addr(&Bsh[nxt][tid * 16]),     (const void*)gn);
      async_copy_b128(lds_addr(&Bsh[nxt][tid * 16 + 8]), (const void*)(gn + 8));
      __builtin_prefetch((const void*)(gn + sliceStride), 0, 3);
    }

    // 16x16 WMMA tile: E(16x128) @ A_i(:,jb:jb+16), K = 4 chunks of 32.
    v8f acc = {0.f, 0.f, 0.f, 0.f, 0.f, 0.f, 0.f, 0.f};
    #pragma unroll
    for (int kc = 0; kc < 4; ++kc) {
      // B operand (32x16): lanes 0-15 hold K={0..7,16..23}, lanes 16-31 hold K={8..15,24..31}.
      const __bf16* q = &Bsh[cur][M * EDIM + kc * 32 + hi * 8];
      B32 tmp; tmp.lo = *(const U4*)q; tmp.hi = *(const U4*)(q + 16);
      v16bf bop = __builtin_bit_cast(v16bf, tmp);
      acc = __builtin_amdgcn_wmma_f32_16x16x32_bf16(false, aop[kc], false, bop,
                                                    (short)0, acc, false, false);
    }

    // f32 scale-accumulate with recurrent state; SshT gives the 8 scale factors
    // contiguously (two b128 LDS reads, feeds v_pk_fma_f32 without shuffles).
    const float* sp = &SshT[i * SPAD + wave * 16 + hi * 8];
    #pragma unroll
    for (int r = 0; r < 8; ++r)
      cacc[r] += sp[r] * acc[r];

    wait_async0();
    __syncthreads();
  }

  #pragma unroll
  for (int r = 0; r < 8; ++r)
    Snext[(size_t)(rowBase + wave * 16 + hi * 8 + r) * RDIM + jb + M] = cacc[r];
}

// ---------------- host ----------------
extern "C" void kernel_launch(void* const* d_in, const int* in_sizes, int n_in,
                              void* d_out, int out_size, void* d_ws, size_t ws_size,
                              hipStream_t stream)
{
  const float* X      = (const float*)d_in[0];
  const float* e1w    = (const float*)d_in[1];
  const float* e1b    = (const float*)d_in[2];
  const float* e2w    = (const float*)d_in[3];
  const float* e2b    = (const float*)d_in[4];
  const float* A      = (const float*)d_in[5];
  const float* init_w = (const float*)d_in[6];
  const float* nw     = (const float*)d_in[7];
  const float* nbv    = (const float*)d_in[8];
  const float* mw     = (const float*)d_in[9];
  const float* mb     = (const float*)d_in[10];
  const float* sw     = (const float*)d_in[11];
  const float* sb     = (const float*)d_in[12];
  const float* aw     = (const float*)d_in[13];
  const float* ab     = (const float*)d_in[14];

  char* w = (char*)d_ws;
  auto carve = [&](size_t bytes) {
    char* p = w; w += (bytes + 255) & ~(size_t)255; return p;
  };
  float*  S0  = (float*)carve((size_t)NB * RDIM * 4);
  float*  S1  = (float*)carve((size_t)NB * RDIM * 4);
  __bf16* Ebf = (__bf16*)carve((size_t)NB * EDIM * 2);
  __bf16* At  = (__bf16*)carve((size_t)RDIM * EDIM * RDIM * 2);

  float* out = (float*)d_out;
  float* rs = out;
  float* ns = out + NB;

  init_kernel<<<(NB * RDIM + 255) / 256, 256, 0, stream>>>(init_w, S0, out);
  transposeA_kernel<<<RDIM * EDIM, RDIM, 0, stream>>>(A, At);

  float* Scur = S0; float* Snxt = S1;
  dim3 ugrid(NB / MTILE, RDIM / 16);   // 128 row-blocks x 8 j-tiles
  for (int t = 0; t < TT; ++t) {
    StepParams p{X, e1w, e1b, e2w, e2b, nw, nbv, mw, mb, sw, sb, aw, ab,
                 Scur, Ebf, rs, ns, t};
    step_kernel<<<NB, 128, 0, stream>>>(p);
    update_kernel<<<ugrid, 128, 0, stream>>>(Ebf, Scur, At, Snxt);
    float* tmp = Scur; Scur = Snxt; Snxt = tmp;
  }
}
